// MemorizingTransformer_80582176407768
// MI455X (gfx1250) — matrix operand order repaired and checked by reading, not compiled, not run on hardware
//
#include <hip/hip_runtime.h>
#include <stdint.h>

typedef __attribute__((ext_vector_type(16))) __bf16  v16bf;
typedef __attribute__((ext_vector_type(8)))  float   v8f;
typedef __attribute__((ext_vector_type(4)))  float   f32x4;
typedef __attribute__((ext_vector_type(2)))  float   f32x2;

#define DEV __device__ __forceinline__

static constexpr int cB = 2, cN = 2048, cDIM = 512, cH = 8, cD = 64, cKR = 32;

DEV v8f wmma_bf16(v16bf a, v16bf b, v8f c) {
  return __builtin_amdgcn_wmma_f32_16x16x32_bf16(false, a, false, b, (short)0, c, false, false);
}

// Low 32 bits of a generic pointer into LDS == LDS byte address (flat aperture
// rule).
DEV unsigned lds_addr32(const void* p) { return (unsigned)(uintptr_t)p; }

// CDNA5 async copy: ASYNCcnt-tracked DMA of 16B/lane straight into LDS,
// no VGPR staging (ISA 15.18.3, GV mode: saddr=off).
DEV void async_copy_b128(unsigned ldsoff, const void* g) {
  asm volatile("global_load_async_to_lds_b128 %0, %1, off"
               :: "v"(ldsoff), "v"((unsigned long long)(uintptr_t)g)
               : "memory");
}
DEV void wait_async0() { asm volatile("s_wait_asynccnt 0x0" ::: "memory"); }

// ---- WMMA fragment builders (contiguous k = 16h + s order, wave32) -------------------
// A 16x32 bf16: lane L: row = L%16, half h = L/16; VGPR j: K = 16h + 2j
DEV v16bf frag_a_f32(const float* A, int lda, int row0, int kb) {
  const int lane = threadIdx.x & 31;
  const int r = lane & 15, h = lane >> 4;
  v16bf a;
#pragma unroll
  for (int j = 0; j < 8; ++j) {
    const int k = kb + 16 * h + 2 * j;
    f32x2 f = *(const f32x2*)(A + (size_t)(row0 + r) * lda + k);
    a[2 * j] = (__bf16)f.x;
    a[2 * j + 1] = (__bf16)f.y;
  }
  return a;
}

DEV v16bf frag_a_bf16(const __bf16* A, int lda, int row0, int kb) {
  const int lane = threadIdx.x & 31;
  const int r = lane & 15, h = lane >> 4;
  v16bf a;
#pragma unroll
  for (int j = 0; j < 8; ++j) {
    const int k = kb + 16 * h + 2 * j;
    const __bf16* p = A + (size_t)(row0 + r) * lda + k;
    a[2 * j] = p[0];
    a[2 * j + 1] = p[1];
  }
  return a;
}

// B 32x16 bf16: lane L: col = L%16, half h = L/16; VGPR j: K = 16h + 2j
DEV v16bf frag_b_f32(const float* W, int ldw, int kb, int col0) {
  const int lane = threadIdx.x & 31;
  const int c = lane & 15, h = lane >> 4;
  v16bf b;
#pragma unroll
  for (int j = 0; j < 8; ++j) {
    const int k = kb + 16 * h + 2 * j;
    b[2 * j]     = (__bf16)W[(size_t)k * ldw + col0 + c];
    b[2 * j + 1] = (__bf16)W[(size_t)(k + 1) * ldw + col0 + c];
  }
  return b;
}

// ---- Generic 16x16-tile WMMA GEMM ------------------------------------------
// MODE 0: q = x@Wq   -> bf16 q[b][h][n][d]
// MODE 1: kv = x@Wkv -> bf16 k[b][n][d], v[b][n][d]
// MODE 2: out = comb(bf16)@Wout + bout -> f32 d_out
template <int MODE>
__global__ __launch_bounds__(128) void gemm_wmma(
    const float* Af, const __bf16* Abf, const float* W, const float* bias,
    __bf16* oq, __bf16* okk, __bf16* ov, float* of,
    int M, int K, int NC, int tilesN, int tilesPerBatch) {
  const int wid = blockIdx.x * 4 + (threadIdx.x >> 5);
  const int lane = threadIdx.x & 31;
  const int b = wid / tilesPerBatch;
  const int tt = wid % tilesPerBatch;
  const int row0 = (tt / tilesN) * 16;
  const int col0 = (tt % tilesN) * 16;

  v8f acc = {};
  for (int kb = 0; kb < K; kb += 32) {
    v16bf a;
    if constexpr (MODE == 2)
      a = frag_a_bf16(Abf, K, row0, kb);
    else
      a = frag_a_f32(Af + (size_t)b * M * K, K, row0, kb);
    v16bf bb = frag_b_f32(W, NC, kb, col0);
    acc = wmma_bf16(a, bb, acc);
  }
  const int c = lane & 15, h = lane >> 4;
#pragma unroll
  for (int j = 0; j < 8; ++j) {
    const int row = row0 + j + 8 * h;   // C frag: VGPR j holds row j+8h
    const int col = col0 + c;
    const float v = acc[j];
    if constexpr (MODE == 0) {
      oq[(((size_t)b * cH + (col >> 6)) * cN + row) * cD + (col & 63)] = (__bf16)v;
    } else if constexpr (MODE == 1) {
      if (col < cD) okk[((size_t)b * cN + row) * cD + col] = (__bf16)v;
      else          ov[((size_t)b * cN + row) * cD + (col - cD)] = (__bf16)v;
    } else {
      of[(size_t)row * NC + col] = v + bias[col];
    }
  }
}

// ---- Local causal flash attention ------------------------------------------
// block = 128 thr (4 waves), each wave owns 16 query rows; key blocks of 32.
// Double-buffered K/V staging: async-to-LDS DMA of block kvb+1 overlaps the
// 8 WMMAs + online softmax of block kvb; one barrier per iteration.
__global__ __launch_bounds__(128) void local_attn(
    const __bf16* qg, const __bf16* kg, const __bf16* vg,
    const float* gate, float* loc) {
  __shared__ __align__(16) __bf16 sK[2][32][72];  // [buf][key][d] (stride 36 dw)
  __shared__ __align__(16) __bf16 sV[2][32][72];  // [buf][key][d]
  __shared__ __align__(16) __bf16 sP[4][16][34];  // per-wave P scratch

  const int qblk = blockIdx.x % (cN / 64);
  const int h = (blockIdx.x / (cN / 64)) % cH;
  const int b = blockIdx.x / ((cN / 64) * cH);
  const int wv = threadIdx.x >> 5;
  const int lane = threadIdx.x & 31;
  const int q0 = qblk * 64 + wv * 16;

  const __bf16* qb = qg + ((size_t)(b * cH + h) * cN) * cD;
  const __bf16* kgb = kg + ((size_t)b * cN) * cD;
  const __bf16* vgb = vg + ((size_t)b * cN) * cD;

  const v16bf aq0 = frag_a_bf16(qb, cD, q0, 0);
  const v16bf aq1 = frag_a_bf16(qb, cD, q0, 32);

  float m[8], l[8];
  v8f o[4];
#pragma unroll
  for (int j = 0; j < 8; ++j) { m[j] = -3.0e38f; l[j] = 0.f; }
#pragma unroll
  for (int t = 0; t < 4; ++t) o[t] = (v8f){};

  const float scale = 0.125f;  // D^-0.5
  const int nkb = 2 * (qblk + 1);

  // per-thread staging slice: 32B of one K row + 32B of one V row
  const int skey = threadIdx.x >> 2;
  const int sdb = (threadIdx.x & 3) * 16;
  unsigned kdst[2], vdst[2];
  kdst[0] = lds_addr32(&sK[0][skey][sdb]);
  kdst[1] = lds_addr32(&sK[1][skey][sdb]);
  vdst[0] = lds_addr32(&sV[0][skey][sdb]);
  vdst[1] = lds_addr32(&sV[1][skey][sdb]);

  // prologue: stage key block 0 into buffer 0
  {
    const __bf16* ksrc = kgb + (size_t)skey * cD + sdb;
    const __bf16* vsrc = vgb + (size_t)skey * cD + sdb;
    async_copy_b128(kdst[0], ksrc);
    async_copy_b128(kdst[0] + 16, ksrc + 8);
    async_copy_b128(vdst[0], vsrc);
    async_copy_b128(vdst[0] + 16, vsrc + 8);
    wait_async0();
  }
  __syncthreads();

  for (int kvb = 0; kvb < nkb; ++kvb) {
    const int key0 = kvb * 32;
    const int cur = kvb & 1;

    // kick off next block's DMA into the other buffer (no wait yet)
    if (kvb + 1 < nkb) {
      const int nxt = cur ^ 1;
      const __bf16* ksrc = kgb + (size_t)(key0 + 32 + skey) * cD + sdb;
      const __bf16* vsrc = vgb + (size_t)(key0 + 32 + skey) * cD + sdb;
      async_copy_b128(kdst[nxt], ksrc);
      async_copy_b128(kdst[nxt] + 16, ksrc + 8);
      async_copy_b128(vdst[nxt], vsrc);
      async_copy_b128(vdst[nxt] + 16, vsrc + 8);
    }

    if (key0 <= q0 + 15) {  // wave-uniform branch: EXEC stays all-ones inside
      const __bf16 (*K)[72] = sK[cur];
      const __bf16 (*V)[72] = sV[cur];
      const int c = lane & 15, hh = lane >> 4;
      v8f s0 = (v8f){}, s1 = (v8f){};
      v16bf bk;
#pragma unroll
      for (int j = 0; j < 8; ++j) { const int d = 16 * hh + 2 * j; bk[2*j] = K[c][d]; bk[2*j+1] = K[c][d+1]; }
      s0 = wmma_bf16(aq0, bk, s0);
#pragma unroll
      for (int j = 0; j < 8; ++j) { const int d = 32 + 16 * hh + 2 * j; bk[2*j] = K[c][d]; bk[2*j+1] = K[c][d+1]; }
      s0 = wmma_bf16(aq1, bk, s0);
#pragma unroll
      for (int j = 0; j < 8; ++j) { const int d = 16 * hh + 2 * j; bk[2*j] = K[16 + c][d]; bk[2*j+1] = K[16 + c][d+1]; }
      s1 = wmma_bf16(aq0, bk, s1);
#pragma unroll
      for (int j = 0; j < 8; ++j) { const int d = 32 + 16 * hh + 2 * j; bk[2*j] = K[16 + c][d]; bk[2*j+1] = K[16 + c][d+1]; }
      s1 = wmma_bf16(aq1, bk, s1);

      const int keyA = key0 + c, keyB = key0 + 16 + c;
      float p0[8], p1[8];
#pragma unroll
      for (int j = 0; j < 8; ++j) {
        const int nrow = q0 + j + 8 * hh;
        float v0 = s0[j] * scale; if (keyA > nrow) v0 = -3.0e38f;
        float v1 = s1[j] * scale; if (keyB > nrow) v1 = -3.0e38f;
        float rm = fmaxf(v0, v1);  // row reduce within 16-lane half = one row
        rm = fmaxf(rm, __shfl_xor(rm, 1, 32));
        rm = fmaxf(rm, __shfl_xor(rm, 2, 32));
        rm = fmaxf(rm, __shfl_xor(rm, 4, 32));
        rm = fmaxf(rm, __shfl_xor(rm, 8, 32));
        const float mn = fmaxf(m[j], rm);
        const float cor = __expf(m[j] - mn);
        const float e0 = __expf(v0 - mn);
        const float e1 = __expf(v1 - mn);
        float rs = e0 + e1;
        rs += __shfl_xor(rs, 1, 32);
        rs += __shfl_xor(rs, 2, 32);
        rs += __shfl_xor(rs, 4, 32);
        rs += __shfl_xor(rs, 8, 32);
        l[j] = l[j] * cor + rs;
        m[j] = mn;
#pragma unroll
        for (int t = 0; t < 4; ++t) o[t][j] *= cor;
        p0[j] = e0;
        p1[j] = e1;
      }
      // P: C-frag -> A-frag relayout via per-wave LDS (same-wave DS in order)
#pragma unroll
      for (int j = 0; j < 8; ++j) {
        sP[wv][j + 8 * hh][c] = (__bf16)p0[j];
        sP[wv][j + 8 * hh][16 + c] = (__bf16)p1[j];
      }
      v16bf ap;
#pragma unroll
      for (int j = 0; j < 8; ++j) {
        const int k = 16 * hh + 2 * j;
        ap[2 * j] = sP[wv][lane & 15][k];
        ap[2 * j + 1] = sP[wv][lane & 15][k + 1];
      }
#pragma unroll
      for (int t = 0; t < 4; ++t) {
        v16bf bv;  // B 32x16: K-dim = key, col = channel (V row-major in LDS)
#pragma unroll
        for (int j = 0; j < 8; ++j) {
          const int kk = 16 * hh + 2 * j;
          bv[2 * j] = V[kk][t * 16 + c];
          bv[2 * j + 1] = V[kk + 1][t * 16 + c];
        }
        o[t] = wmma_bf16(ap, bv, o[t]);
      }
    }

    // my DMA for kvb+1 landed; everyone done copying AND done reading buf[cur]
    wait_async0();
    __syncthreads();
  }

  const int c = lane & 15, hh = lane >> 4;
  const float g = 1.f / (1.f + __expf(-gate[h]));
#pragma unroll
  for (int t = 0; t < 4; ++t)
#pragma unroll
    for (int j = 0; j < 8; ++j) {
      const int nrow = q0 + j + 8 * hh;
      loc[((size_t)(b * cH + h) * cN + nrow) * cD + t * 16 + c] = g * o[t][j] / l[j];
    }
}

// ---- Memory-retrieval attention: pure HBM streamer (537 MB, NT loads) ------
__global__ __launch_bounds__(256) void mem_attn(
    const __bf16* qg, const float* mk, const float* mv,
    const unsigned char* mask, const float* nk, const float* nv,
    const float* gate, const float* loc, __bf16* comb) {
  __shared__ __align__(16) float qsh[8][64];
  const int wv = threadIdx.x >> 5;
  const int lane = threadIdx.x & 31;
  const int wid = blockIdx.x * 8 + wv;
  const int n = wid % cN;
  const int h = (wid / cN) % cH;
  const int b = wid / (cN * cH);
  const size_t qidx = (size_t)(b * cH + h) * cN + n;

  // get the V stream in flight early (global_prefetch_b8)
  const float* vbase = mv + qidx * cKR * cD;
  __builtin_prefetch(vbase + (size_t)lane * cD, 0, 1);

  const __bf16* qrow = qg + qidx * cD;
  qsh[wv][2 * lane] = (float)qrow[2 * lane];
  qsh[wv][2 * lane + 1] = (float)qrow[2 * lane + 1];
  __syncthreads();

  // phase 1: lane = retrieved key j; 32 lanes stream 8 KB contiguous block
  const float* kr = mk + (qidx * cKR + lane) * cD;
  const f32x4* q4 = (const f32x4*)&qsh[wv][0];
  float s = 0.f;
#pragma unroll
  for (int t = 0; t < 16; ++t) {
    const f32x4 kk = __builtin_nontemporal_load((const f32x4*)(kr + 4 * t));
    const f32x4 qq = q4[t];
    s += qq.x * kk.x + qq.y * kk.y + qq.z * kk.z + qq.w * kk.w;
  }
  s *= 0.125f;
  const bool ok = mask[qidx * cKR + lane] != 0;
  const float sm = ok ? s : -3.0e38f;

  float sn = 0.f;  // null key (uniform broadcast loads, L2-resident)
  const f32x4* nk4 = (const f32x4*)nk;
#pragma unroll
  for (int t = 0; t < 16; ++t) {
    const f32x4 kk = nk4[t];
    const f32x4 qq = q4[t];
    sn += qq.x * kk.x + qq.y * kk.y + qq.z * kk.z + qq.w * kk.w;
  }
  sn *= 0.125f;

  float mx = sm;
  mx = fmaxf(mx, __shfl_xor(mx, 1, 32));
  mx = fmaxf(mx, __shfl_xor(mx, 2, 32));
  mx = fmaxf(mx, __shfl_xor(mx, 4, 32));
  mx = fmaxf(mx, __shfl_xor(mx, 8, 32));
  mx = fmaxf(mx, __shfl_xor(mx, 16, 32));
  mx = fmaxf(mx, sn);
  const float p = ok ? __expf(s - mx) : 0.f;
  const float pn = __expf(sn - mx);
  float sum = p;
  sum += __shfl_xor(sum, 1, 32);
  sum += __shfl_xor(sum, 2, 32);
  sum += __shfl_xor(sum, 4, 32);
  sum += __shfl_xor(sum, 8, 32);
  sum += __shfl_xor(sum, 16, 32);
  sum += pn;
  const float inv = 1.f / sum;

  // phase 2: lane = channel pair; p_j broadcast by shuffle; coalesced V rows
  const int c0 = 2 * lane;
  float a0 = pn * nv[c0], a1 = pn * nv[c0 + 1];
#pragma unroll 8
  for (int j = 0; j < cKR; ++j) {
    const float pj = __shfl(p, j, 32);
    const f32x2 vv = __builtin_nontemporal_load((const f32x2*)(vbase + (size_t)j * cD + c0));
    a0 += pj * vv.x;
    a1 += pj * vv.y;
  }
  a0 *= inv;
  a1 *= inv;

  const float g = 1.f / (1.f + __expf(-gate[h]));
  const float o0 = loc[qidx * cD + c0] + (1.f - g) * a0;       // loc = g*local
  const float o1 = loc[qidx * cD + c0 + 1] + (1.f - g) * a1;
  union { unsigned u; __bf16 hv[2]; } pk;
  pk.hv[0] = (__bf16)o0;
  pk.hv[1] = (__bf16)o1;
  *(unsigned*)(comb + ((size_t)b * cN + n) * cDIM + h * cD + c0) = pk.u;
}

// ---- Launch -----------------------------------------------------------------
extern "C" void kernel_launch(void* const* d_in, const int* in_sizes, int n_in,
                              void* d_out, int out_size, void* d_ws, size_t ws_size,
                              hipStream_t stream) {
  (void)in_sizes; (void)n_in; (void)out_size; (void)ws_size;
  const float* x = (const float*)d_in[0];
  const float* mem_k = (const float*)d_in[1];
  const float* mem_v = (const float*)d_in[2];
  const unsigned char* mem_mask = (const unsigned char*)d_in[3];
  const float* Wq = (const float*)d_in[4];
  const float* Wkv = (const float*)d_in[5];
  const float* Wout = (const float*)d_in[6];
  const float* bout = (const float*)d_in[7];
  const float* null_k = (const float*)d_in[8];
  const float* null_v = (const float*)d_in[9];
  const float* gate = (const float*)d_in[10];

  char* ws = (char*)d_ws;
  __bf16* q_bf = (__bf16*)ws;                                   // 4 MB
  __bf16* k_bf = (__bf16*)(ws + (4u << 20));                    // 512 KB
  __bf16* v_bf = (__bf16*)(ws + (4u << 20) + (512u << 10));     // 512 KB
  float* loc = (float*)(ws + (5u << 20));                       // 8 MB
  __bf16* comb = (__bf16*)(ws + (13u << 20));                   // 4 MB

  // q projection: B * (2048/16)*(512/16) = 8192 wave-tiles
  gemm_wmma<0><<<2048, 128, 0, stream>>>(x, nullptr, Wq, nullptr,
                                         q_bf, nullptr, nullptr, nullptr,
                                         cN, cDIM, cH * cD, 32, 4096);
  // kv projection: B * 128*8 = 2048 wave-tiles
  gemm_wmma<1><<<512, 128, 0, stream>>>(x, nullptr, Wkv, nullptr,
                                        nullptr, k_bf, v_bf, nullptr,
                                        cN, cDIM, 2 * cD, 8, 1024);
  // local causal attention: B*H*(N/64) blocks
  local_attn<<<cB * cH * (cN / 64), 128, 0, stream>>>(q_bf, k_bf, v_bf, gate, loc);
  // memory attention + gated combine: B*H*N waves / 8 per block
  mem_attn<<<cB * cH * cN / 8, 256, 0, stream>>>(q_bf, mem_k, mem_v, mem_mask,
                                                 null_k, null_v, gate, loc, comb);
  // out projection: (B*N/16)*(512/16) = 8192 wave-tiles
  gemm_wmma<2><<<2048, 128, 0, stream>>>(nullptr, comb, Wout, bout,
                                         nullptr, nullptr, nullptr, (float*)d_out,
                                         cB * cN, cDIM, cDIM, 32, 8192);
}
